// selfAttention_55035710931797
// MI455X (gfx1250) — matrix operand order repaired and checked by reading, not compiled, not run on hardware
//
#include <hip/hip_runtime.h>
#include <hip/hip_bf16.h>
#include <math.h>

// ---------------------------------------------------------------------------
// Self-attention forward, MI455X (gfx1250), wave32 + v_wmma_f32_16x16x32_bf16
// B=2, T=2048, C=1024, H=16, D=64
// ---------------------------------------------------------------------------

typedef __attribute__((ext_vector_type(16))) __bf16 v16bf;
typedef __attribute__((ext_vector_type(8)))  float  v8f;

union BF16Frag {
    v16bf v;
    uint4 q[2];
};

__device__ __forceinline__ unsigned short f2bf(float f) {
    unsigned int u = __float_as_uint(f);
    unsigned int r = (u + 0x7FFFu + ((u >> 16) & 1u)) >> 16;  // round-nearest-even
    return (unsigned short)r;
}

__device__ __forceinline__ v8f wmma_bf16(const BF16Frag& a, const BF16Frag& b, v8f c) {
    return __builtin_amdgcn_wmma_f32_16x16x32_bf16(
        /*neg_a=*/false, a.v, /*neg_b=*/false, b.v,
        /*c_mod=*/(short)0, c, /*reuse_a=*/false, /*reuse_b=*/false);
}

// ------------------------- conversion kernels ------------------------------

__global__ __launch_bounds__(256) void cvt_bf16(const float* __restrict__ in,
                                                unsigned short* __restrict__ out, int n) {
    int i = blockIdx.x * blockDim.x + threadIdx.x;
    if (i < n) out[i] = f2bf(in[i]);
}

// 1024x1024 transpose + convert: out[c][r] = in[r][c]
__global__ __launch_bounds__(256) void cvt_bf16_t(const float* __restrict__ in,
                                                  unsigned short* __restrict__ out) {
    int i = blockIdx.x * blockDim.x + threadIdx.x;  // 0 .. 1M-1
    int r = i >> 10;
    int c = i & 1023;
    out[c * 1024 + r] = f2bf(in[r * 1024 + c]);
}

// ------------------------------ GEMM ---------------------------------------
// A  : [4096, 1024] bf16 row-major
// Bt : [1024, 1024] bf16 (transposed weight: Bt[n][k] = W[k][n])
// Wave computes a 32x64 output tile (2 row-tiles x 4 col-tiles): each B
// fragment feeds two WMMAs -> 12 b128 loads per 8 WMMAs.
// mode 0: store bf16 to [B,H,T,D]   (Q, K)
// mode 1: store bf16 to [B,H,D,T]   (V transposed)
// mode 2: store fp32 to [4096,1024] (final projection -> d_out)
__global__ __launch_bounds__(256) void gemm_bf16(const unsigned short* __restrict__ A,
                                                 const unsigned short* __restrict__ Bt,
                                                 const float* __restrict__ bias,
                                                 void* __restrict__ out, int mode) {
    const int Kdim = 1024;
    int wave = blockIdx.x * (blockDim.x >> 5) + (threadIdx.x >> 5);
    int lane = threadIdx.x & 31;
    int lm = lane & 15;
    int hi = lane >> 4;

    int rowblk = wave & 127;   // 4096/32 row blocks
    int colblk = wave >> 7;    // 1024/64 col blocks
    int row0 = rowblk * 32;
    int col0 = colblk * 64;

    v8f acc[2][4];
    for (int m = 0; m < 2; ++m)
        for (int n = 0; n < 4; ++n)
            for (int j = 0; j < 8; ++j) acc[m][n][j] = 0.0f;

    const unsigned short* arow0 = A + (size_t)(row0 + lm) * Kdim;
    const unsigned short* arow1 = A + (size_t)(row0 + 16 + lm) * Kdim;

    for (int k0 = 0; k0 < Kdim; k0 += 32) {
        BF16Frag a0, a1;
        // A layout: lanes 0-15 hold K = k0+0..7 and k0+16..23; lanes 16-31 +8
        a0.q[0] = *(const uint4*)(arow0 + k0 + hi * 8);
        a0.q[1] = *(const uint4*)(arow0 + k0 + 16 + hi * 8);
        a1.q[0] = *(const uint4*)(arow1 + k0 + hi * 8);
        a1.q[1] = *(const uint4*)(arow1 + k0 + 16 + hi * 8);
        for (int n = 0; n < 4; ++n) {
            BF16Frag b;
            // B layout: lane = column, K = k0+0..7 and k0+16..23; lanes 16-31 +8
            const unsigned short* bcol =
                Bt + (size_t)(col0 + n * 16 + lm) * Kdim + k0 + hi * 8;
            b.q[0] = *(const uint4*)(bcol);
            b.q[1] = *(const uint4*)(bcol + 16);
            acc[0][n] = wmma_bf16(a0, b, acc[0][n]);
            acc[1][n] = wmma_bf16(a1, b, acc[1][n]);
        }
    }

    for (int m = 0; m < 2; ++m) {
        for (int n = 0; n < 4; ++n) {
            for (int r = 0; r < 8; ++r) {
                int gr = row0 + m * 16 + r + 8 * hi;  // 0..4095 (= b*T + t)
                int gc = col0 + n * 16 + lm;          // 0..1023 (= h*64 + d)
                float val = acc[m][n][r] + bias[gc];
                if (mode == 2) {
                    ((float*)out)[(size_t)gr * 1024 + gc] = val;
                } else {
                    int b = gr >> 11, t = gr & 2047;
                    int h = gc >> 6,  d = gc & 63;
                    unsigned short bf = f2bf(val);
                    if (mode == 0)
                        ((unsigned short*)out)[(((size_t)(b * 16 + h) * 2048) + t) * 64 + d] = bf;
                    else
                        ((unsigned short*)out)[(((size_t)(b * 16 + h) * 64) + d) * 2048 + t] = bf;
                }
            }
        }
    }
}

// --------------------------- flash attention -------------------------------
// Q : [B,H,T,D] bf16   K : [B,H,T,D] bf16   Vt : [B,H,D,T] bf16
// Yc: [B,T,C]   bf16   (context, C index = h*64 + d)
// LDS P-tile round trip is wave-private: CDNA5 DS ops from one wave are
// in-order, so a compiler scheduling fence (wave_barrier) suffices — no
// s_barrier in the hot loop.
__global__ __launch_bounds__(256) void attn_kernel(const unsigned short* __restrict__ Q,
                                                   const unsigned short* __restrict__ K,
                                                   const unsigned short* __restrict__ Vt,
                                                   unsigned short* __restrict__ Yc) {
    const int T = 2048, D = 64;
    __shared__ unsigned short Pl[8][16 * 32];   // per-wave P tile staging

    int wslot = threadIdx.x >> 5;
    int wave = blockIdx.x * 8 + wslot;
    int lane = threadIdx.x & 31;
    int lm = lane & 15;
    int hi = lane >> 4;

    int qblk = wave & 127;   // T/16 = 128 query blocks
    int bh   = wave >> 7;    // 0..31 (b*H + h)

    const unsigned short* Qp = Q  + (size_t)bh * T * D;
    const unsigned short* Kp = K  + (size_t)bh * T * D;
    const unsigned short* Vp = Vt + (size_t)bh * D * T;

    // Q fragments for the whole row block: 2 K-steps of 32 over D=64
    BF16Frag qf[2];
    const unsigned short* qrow = Qp + (size_t)(qblk * 16 + lm) * D;
    for (int s = 0; s < 2; ++s) {
        qf[s].q[0] = *(const uint4*)(qrow + s * 32 + hi * 8);
        qf[s].q[1] = *(const uint4*)(qrow + s * 32 + 16 + hi * 8);
    }

    v8f o[4];
    for (int n = 0; n < 4; ++n)
        for (int j = 0; j < 8; ++j) o[n][j] = 0.0f;
    float m[8], l[8];
    for (int r = 0; r < 8; ++r) { m[r] = -INFINITY; l[r] = 0.0f; }

    const float scale = 0.125f;  // 1/sqrt(64)

    for (int kb = 0; kb < T; kb += 32) {
        // ---- S = Q * K^T for 32 keys (two 16x16 tiles) ----
        v8f s[2];
        for (int t2 = 0; t2 < 2; ++t2)
            for (int j = 0; j < 8; ++j) s[t2][j] = 0.0f;
        for (int sub = 0; sub < 2; ++sub) {
            int keyb = kb + sub * 16;
            for (int ds = 0; ds < 2; ++ds) {
                BF16Frag kf;
                const unsigned short* kp =
                    Kp + (size_t)(keyb + lm) * D + ds * 32 + hi * 8;
                kf.q[0] = *(const uint4*)(kp);
                kf.q[1] = *(const uint4*)(kp + 16);
                s[sub] = wmma_bf16(qf[ds], kf, s[sub]);
            }
        }

        // ---- online softmax (rows live across one 16-lane half) ----
        for (int r = 0; r < 8; ++r) {
            float x0 = s[0][r] * scale;
            float x1 = s[1][r] * scale;
            float tm = fmaxf(x0, x1);
            for (int off = 8; off >= 1; off >>= 1)
                tm = fmaxf(tm, __shfl_xor(tm, off, 16));
            float mn = fmaxf(m[r], tm);
            float alpha = __expf(m[r] - mn);
            float p0 = __expf(x0 - mn);
            float p1 = __expf(x1 - mn);
            float ts = p0 + p1;
            for (int off = 8; off >= 1; off >>= 1)
                ts += __shfl_xor(ts, off, 16);
            l[r] = l[r] * alpha + ts;
            m[r] = mn;
            for (int n = 0; n < 4; ++n) o[n][r] *= alpha;
            // stash P (bf16) in LDS, row-major [16 rows x 32 keys]
            int row = r + 8 * hi;
            Pl[wslot][row * 32 + lm]      = f2bf(p0);
            Pl[wslot][row * 32 + 16 + lm] = f2bf(p1);
        }
        // stores above must issue before the reads below; DS is in-order
        // within a wave, so a scheduling fence is sufficient.
        __builtin_amdgcn_wave_barrier();

        // ---- O += P * V ----
        BF16Frag pf;
        const unsigned short* prow = &Pl[wslot][lm * 32];
        pf.q[0] = *(const uint4*)(prow + hi * 8);
        pf.q[1] = *(const uint4*)(prow + 16 + hi * 8);
        for (int n = 0; n < 4; ++n) {
            BF16Frag vf;
            const unsigned short* vp =
                Vp + (size_t)(n * 16 + lm) * T + kb + hi * 8;
            vf.q[0] = *(const uint4*)(vp);
            vf.q[1] = *(const uint4*)(vp + 16);
            o[n] = wmma_bf16(pf, vf, o[n]);
        }
        // keep next iteration's P stores after this iteration's P reads
        __builtin_amdgcn_wave_barrier();
    }

    // ---- normalize and store context ----
    int b = bh >> 4, h = bh & 15;
    for (int n = 0; n < 4; ++n) {
        for (int r = 0; r < 8; ++r) {
            int t = qblk * 16 + r + 8 * hi;
            int d = n * 16 + lm;
            float val = o[n][r] / l[r];
            Yc[((size_t)(b * 2048 + t)) * 1024 + h * 64 + d] = f2bf(val);
        }
    }
}

// ------------------------------ launcher -----------------------------------

extern "C" void kernel_launch(void* const* d_in, const int* in_sizes, int n_in,
                              void* d_out, int out_size, void* d_ws, size_t ws_size,
                              hipStream_t stream) {
    const float* x  = (const float*)d_in[0];
    const float* Wk = (const float*)d_in[1];
    const float* bk = (const float*)d_in[2];
    const float* Wq = (const float*)d_in[3];
    const float* bq = (const float*)d_in[4];
    const float* Wv = (const float*)d_in[5];
    const float* bv = (const float*)d_in[6];
    const float* Wp = (const float*)d_in[7];
    const float* bp = (const float*)d_in[8];
    float* out = (float*)d_out;

    // workspace carve-up (bf16 buffers), total 48 MiB
    char* ws = (char*)d_ws;
    unsigned short* xb  = (unsigned short*)ws; ws += (size_t)4096 * 1024 * 2;
    unsigned short* Wqt = (unsigned short*)ws; ws += (size_t)1024 * 1024 * 2;
    unsigned short* Wkt = (unsigned short*)ws; ws += (size_t)1024 * 1024 * 2;
    unsigned short* Wvt = (unsigned short*)ws; ws += (size_t)1024 * 1024 * 2;
    unsigned short* Wpt = (unsigned short*)ws; ws += (size_t)1024 * 1024 * 2;
    unsigned short* Qb  = (unsigned short*)ws; ws += (size_t)4096 * 1024 * 2;
    unsigned short* Kb  = (unsigned short*)ws; ws += (size_t)4096 * 1024 * 2;
    unsigned short* Vtb = (unsigned short*)ws; ws += (size_t)4096 * 1024 * 2;
    unsigned short* Yc  = (unsigned short*)ws; ws += (size_t)4096 * 1024 * 2;

    // 1) fp32 -> bf16 staging
    cvt_bf16<<<(4096 * 1024) / 256, 256, 0, stream>>>(x, xb, 4096 * 1024);
    cvt_bf16_t<<<(1024 * 1024) / 256, 256, 0, stream>>>(Wq, Wqt);
    cvt_bf16_t<<<(1024 * 1024) / 256, 256, 0, stream>>>(Wk, Wkt);
    cvt_bf16_t<<<(1024 * 1024) / 256, 256, 0, stream>>>(Wv, Wvt);
    cvt_bf16_t<<<(1024 * 1024) / 256, 256, 0, stream>>>(Wp, Wpt);

    // 2) QKV projections (2048 waves each: 256 blocks x 8 waves, 32x64 tiles)
    gemm_bf16<<<256, 256, 0, stream>>>(xb, Wqt, bq, (void*)Qb, 0);
    gemm_bf16<<<256, 256, 0, stream>>>(xb, Wkt, bk, (void*)Kb, 0);
    gemm_bf16<<<256, 256, 0, stream>>>(xb, Wvt, bv, (void*)Vtb, 1);

    // 3) flash attention: 4096 waves (B*H * T/16)
    attn_kernel<<<512, 256, 0, stream>>>(Qb, Kb, Vtb, Yc);

    // 4) output projection -> fp32 d_out
    gemm_bf16<<<256, 256, 0, stream>>>(Yc, Wpt, bp, (void*)out, 2);
}
